// BottomUpHTMM_61349312856744
// MI455X (gfx1250) — matrix operand order repaired and checked by reading, not compiled, not run on hardware
//
#include <hip/hip_runtime.h>
#include <math.h>

// ---- problem constants (match reference) ----
#define N_TREES 128
#define DEPTH   10
#define C_DIM   16
#define M_DIM   256
#define L_DIM   2
#define G_DIM   8
#define NODES_PER_TREE 1023   // 2^10 - 1

typedef __attribute__((ext_vector_type(16))) _Float16 v16h;
typedef __attribute__((ext_vector_type(8)))  _Float16 v8h;
typedef __attribute__((ext_vector_type(8)))  float    v8f;

// Workspace layout (bytes):
//   W   : f16 [g][pos][a][b]          = 8*2*16*16*2  = 8192 B
//   Bn  : f32 [g][m][c]               = 8*256*16*4   = 131072 B
//   PiN : f32 [g][pos][c]             = 8*2*16*4     = 1024 B
#define WS_W_OFF   0
#define WS_BN_OFF  8192
#define WS_PI_OFF  (8192 + 131072)

// ds_swizzle offset: group-of-32, xor_mask=0x10, or=0, and=0x1f  (SWAPX16)
#define SWZ_XOR16  0x401F

__device__ __forceinline__ float fast_rcp(float x) {
  return __builtin_amdgcn_rcpf(x);   // single v_rcp_f32 (x > 0 here)
}
__device__ __forceinline__ float fast_ln(float x) {
  // raw v_log_f32 (log2) * ln(2); nu is a normalized probability mass,
  // never denormal, so skip the ldexp/denorm-safe expansion of __logf.
  return __builtin_amdgcn_logf(x) * 0.69314718055994531f;
}

// ---------------------------------------------------------------------------
// Prep: softmaxes of lambda_A (over a), lambda_B (over m), lambda_Pi (over c),
// lambda_SP (over l); fold SP into the transition matrices, store f16 W.
// ---------------------------------------------------------------------------
__global__ __launch_bounds__(256)
void htmm_prep(const float* __restrict__ lA, const float* __restrict__ lB,
               const float* __restrict__ lPi, const float* __restrict__ lSP,
               _Float16* __restrict__ W, float* __restrict__ Bn,
               float* __restrict__ PiN) {
  const int tid = threadIdx.x;

  // ---- A softmax over 'a' for each (b,l,g), times softmax(SP)[l,g] ----
  {
    const int g = tid & 7;
    const int l = (tid >> 3) & 1;
    const int b = tid >> 4;               // 0..15
    float s0 = lSP[0 * G_DIM + g], s1 = lSP[1 * G_DIM + g];
    float sm = fmaxf(s0, s1);
    float e0 = __expf(s0 - sm), e1 = __expf(s1 - sm);
    float sp = ((l == 0) ? e0 : e1) * fast_rcp(e0 + e1);

    float mx = -1e30f;
    float va[C_DIM];
#pragma unroll
    for (int a = 0; a < C_DIM; ++a) {
      va[a] = lA[(((a * C_DIM + b) * L_DIM) + l) * G_DIM + g];
      mx = fmaxf(mx, va[a]);
    }
    float sum = 0.f;
#pragma unroll
    for (int a = 0; a < C_DIM; ++a) { va[a] = __expf(va[a] - mx); sum += va[a]; }
    float inv = sp * fast_rcp(sum);
#pragma unroll
    for (int a = 0; a < C_DIM; ++a)
      W[(((g * 2 + l) * C_DIM) + a) * C_DIM + b] = (_Float16)(va[a] * inv);
  }

  // ---- B softmax over m for each (c,g); store Bn[g][m][c] ----
  if (tid < C_DIM * G_DIM) {
    const int c = tid >> 3, g = tid & 7;
    float mx = -1e30f;
    for (int m = 0; m < M_DIM; ++m)
      mx = fmaxf(mx, lB[(c * M_DIM + m) * G_DIM + g]);
    float sum = 0.f;
    for (int m = 0; m < M_DIM; ++m)
      sum += __expf(lB[(c * M_DIM + m) * G_DIM + g] - mx);
    float inv = fast_rcp(sum);
    for (int m = 0; m < M_DIM; ++m)
      Bn[(g * M_DIM + m) * C_DIM + c] =
          __expf(lB[(c * M_DIM + m) * G_DIM + g] - mx) * inv;
  }

  // ---- Pi softmax over c for each (l,g); store PiN[g][l][c] ----
  if (tid < L_DIM * G_DIM) {
    const int l = tid & 1, g = tid >> 1;
    float mx = -1e30f;
#pragma unroll
    for (int c = 0; c < C_DIM; ++c)
      mx = fmaxf(mx, lPi[(c * L_DIM + l) * G_DIM + g]);
    float sum = 0.f;
    float v[C_DIM];
#pragma unroll
    for (int c = 0; c < C_DIM; ++c) {
      v[c] = __expf(lPi[(c * L_DIM + l) * G_DIM + g] - mx);
      sum += v[c];
    }
    float inv = fast_rcp(sum);
#pragma unroll
    for (int c = 0; c < C_DIM; ++c)
      PiN[(g * L_DIM + l) * C_DIM + c] = v[c] * inv;
  }
}

// ---------------------------------------------------------------------------
// Main: one block per (tree, gen). Whole tree state held in LDS as f16.
// Per 16-parent tile: one v_wmma_f32_16x16x32_f16 (K=0..15 <- W_pos0 x betaL,
// K=16..31 <- W_pos1 x betaR) immediately followed by in-register
// normalization: lane(n,h) holds rows 8h..8h+7 of parent n; partner-lane sum
// via ds_swizzle SWAPX16 gives nu; scaled f16 beta written for next level.
// One barrier per level.
// ---------------------------------------------------------------------------
__global__ __launch_bounds__(256)
void htmm_main(const int* __restrict__ x, const _Float16* __restrict__ W,
               const float* __restrict__ Bn, const float* __restrict__ PiN,
               float* __restrict__ out) {
  // f16 beta ping/pong: [0..512*16) and [512*16..512*16+256*16)
  __shared__ __align__(32) _Float16 bufh[(512 + 256) * 16];
  __shared__ float red[256];

  const int tid  = threadIdx.x;
  const int lane = tid & 31;
  const int m    = lane & 15;   // A-row / D-column / parent-in-tile index
  const int h    = lane >> 4;   // K-half & D-row-half selector
  const int wave = tid >> 5;
  const int blk  = blockIdx.x;
  const int t    = blk >> 3;    // tree
  const int g    = blk & 7;     // generator
  const int base = t * NODES_PER_TREE;

  float llacc = 0.f;

  // ---- preload A operand (per-lane, invariant over levels) ----
  // lane(m,h): elems 0..15 = W_pos_h[m][0..15]
  v16h avec;
#pragma unroll
  for (int e = 0; e < 8; ++e) {
    avec[e]     = W[(((g * 2 + h) * C_DIM) + m) * C_DIM + e];
    avec[e + 8] = W[(((g * 2 + h) * C_DIM) + m) * C_DIM + (8 + e)];
  }

  // ---- leaves: beta = Pi[:,pos] * B[:,x], normalized; ll += log(nu) ----
  for (int j = tid; j < 512; j += 256) {
    const int node = 511 + j;              // local node index
    const int xv   = x[base + node];
    const int pos  = j & 1;
    const float* bn = &Bn[(g * M_DIM + xv) * C_DIM];
    const float* pi = &PiN[(g * L_DIM + pos) * C_DIM];
    float v[C_DIM];
#pragma unroll
    for (int c = 0; c < C_DIM; ++c) v[c] = pi[c] * bn[c];
    // tree-structured sum (depth 4 instead of 16)
    float s[8];
#pragma unroll
    for (int c = 0; c < 8; ++c) s[c] = v[2 * c] + v[2 * c + 1];
    float q0 = (s[0] + s[1]) + (s[2] + s[3]);
    float q1 = (s[4] + s[5]) + (s[6] + s[7]);
    const float nu  = q0 + q1;
    const float inv = fast_rcp(nu);
    v16h hv;
#pragma unroll
    for (int c = 0; c < C_DIM; ++c) hv[c] = (_Float16)(v[c] * inv);
    *(v16h*)&bufh[j * 16] = hv;
    llacc += fast_ln(nu);
  }
  __syncthreads();

  // ---- bottom-up over levels k = 9..1 (one barrier per level) ----
  int curOff = 0;            // children beta live here (f16)
  int nxtOff = 512 * 16;     // parents written here   (f16)
  for (int k = DEPTH - 1; k >= 1; --k) {
    const int nch    = 1 << k;
    const int np     = nch >> 1;
    const int ntiles = (np + 15) >> 4;

    for (int tile = wave; tile < ntiles; tile += 8) {   // wave-uniform loop
      const int p0 = tile << 4;
      int j = 2 * (p0 + m) + h;                          // child level-local
      if (j > nch - 1) j = nch - 1;                      // clamp padding lanes
      const v16h bvec = *(const v16h*)&bufh[curOff + j * 16];
      v8f acc = {};
      // D[a][n] = sum_b W0[a][b]*betaL[b][n] + W1[a][b]*betaR[b][n]
      acc = __builtin_amdgcn_wmma_f32_16x16x32_f16(
          false, avec, false, bvec, (short)0, acc, false, false);

      // fused normalization: lane(n=m, h) owns rows a = 8h..8h+7 of parent n
      const int p  = p0 + m;
      const int pc = (p < np) ? p : (np - 1);
      const int u  = (np - 1) + pc;                      // parent local node
      const int xv = x[base + u];
      const float* bn = &Bn[(g * M_DIM + xv) * C_DIM + 8 * h];
      float v[8];
#pragma unroll
      for (int r = 0; r < 8; ++r) v[r] = acc[r] * bn[r];
      // tree-structured partial sum (depth 3 instead of 8)
      const float p01 = v[0] + v[1], p23 = v[2] + v[3];
      const float p45 = v[4] + v[5], p67 = v[6] + v[7];
      const float part = (p01 + p23) + (p45 + p67);
      const float other = __uint_as_float(
          __builtin_amdgcn_ds_swizzle(__float_as_uint(part), SWZ_XOR16));
      const float nu  = part + other;
      const float inv = fast_rcp(nu);
      v8h hv;
#pragma unroll
      for (int r = 0; r < 8; ++r) hv[r] = (_Float16)(v[r] * inv);
      if (p < np) {
        *(v8h*)&bufh[nxtOff + p * 16 + 8 * h] = hv;      // finished f16 beta
        if (h == 0) llacc += fast_ln(nu);                // one count per node
      }
    }
    __syncthreads();
    const int tmp = curOff; curOff = nxtOff; nxtOff = tmp;
  }

  // ---- block reduction of per-thread log-likelihood ----
  red[tid] = llacc;
  __syncthreads();
  for (int s = 128; s > 0; s >>= 1) {
    if (tid < s) red[tid] += red[tid + s];
    __syncthreads();
  }
  if (tid == 0) out[t * G_DIM + g] = red[0];
}

// ---------------------------------------------------------------------------
extern "C" void kernel_launch(void* const* d_in, const int* in_sizes, int n_in,
                              void* d_out, int out_size, void* d_ws, size_t ws_size,
                              hipStream_t stream) {
  const float* lA  = (const float*)d_in[0];  // (16,16,2,8)
  const float* lB  = (const float*)d_in[1];  // (16,256,8)
  const float* lPi = (const float*)d_in[2];  // (16,2,8)
  const float* lSP = (const float*)d_in[3];  // (2,8)
  const int*   x   = (const int*)d_in[4];    // (130944,)
  float* out = (float*)d_out;                // (128,8)

  char* ws = (char*)d_ws;
  _Float16* W  = (_Float16*)(ws + WS_W_OFF);
  float*    Bn = (float*)(ws + WS_BN_OFF);
  float*    Pi = (float*)(ws + WS_PI_OFF);

  htmm_prep<<<1, 256, 0, stream>>>(lA, lB, lPi, lSP, W, Bn, Pi);
  htmm_main<<<N_TREES * G_DIM, 256, 0, stream>>>(x, W, Bn, Pi, out);
}
